// TritonFusedMoeLinearBase_76647986365154
// MI455X (gfx1250) — matrix operand order repaired and checked by reading, not compiled, not run on hardware
//
#include <hip/hip_runtime.h>
#include <hip/hip_bf16.h>

typedef __attribute__((ext_vector_type(16))) __bf16        v16bf;
typedef __attribute__((ext_vector_type(8)))  float         v8f;
typedef __attribute__((ext_vector_type(4)))  float         v4f;
typedef __attribute__((ext_vector_type(4)))  unsigned int  v4u;
typedef __attribute__((ext_vector_type(2)))  unsigned int  v2u;

#define NUM_TOKENS   8192
#define IN_FEAT      2048
#define OUT_FEAT     8192
#define NUM_EXPERTS  8
#define BM           128
#define BN           128
#define BK           32
#define KSTEPS       (IN_FEAT / BK)   // 64
#define LDSTR        40               // ushort stride per row: 32 data + 8 pad (80B -> conflict-free b128)
#define MAX_MTILES   (NUM_TOKENS / BM + NUM_EXPERTS - 1)  // 71

// ---------- fp32 pair -> packed bf16 (round-half-up; differs from RNE only on exact ties)
// 2x v_add_nc_u32 + 1x v_perm_b32 per 2 elements.
__device__ __forceinline__ unsigned int pk2(float a, float b) {
    unsigned int ua = __float_as_uint(a) + 0x8000u;
    unsigned int ub = __float_as_uint(b) + 0x8000u;
    // v_perm_b32: selector bytes index {S0:S1} (S1 = bytes 0-3, S0 = bytes 4-7).
    // D = { ub[31:16], ua[31:16] }  -> bytes {7,6,3,2}
    return __builtin_amdgcn_perm(ub, ua, 0x07060302u);
}

// ---------- Kernel 1: bucket tokens by expert ----------
// ws layout: int perm[NUM_TOKENS]; int offs[NUM_EXPERTS+1];
__global__ __launch_bounds__(1024)
void moe_sort_kernel(const int* __restrict__ eids, int* __restrict__ perm,
                     int* __restrict__ offs) {
    __shared__ int cnt[NUM_EXPERTS];
    __shared__ int cur[NUM_EXPERTS];
    __shared__ int off_sh[NUM_EXPERTS + 1];
    const int tid = threadIdx.x;
    if (tid < NUM_EXPERTS) cnt[tid] = 0;
    __syncthreads();
    for (int t = tid; t < NUM_TOKENS; t += 1024)
        atomicAdd(&cnt[eids[t] & (NUM_EXPERTS - 1)], 1);
    __syncthreads();
    if (tid == 0) {
        int s = 0;
        for (int e = 0; e < NUM_EXPERTS; ++e) { off_sh[e] = s; cur[e] = s; s += cnt[e]; }
        off_sh[NUM_EXPERTS] = s;
    }
    __syncthreads();
    for (int t = tid; t < NUM_TOKENS; t += 1024) {
        int e = eids[t] & (NUM_EXPERTS - 1);
        int p = atomicAdd(&cur[e], 1);
        perm[p] = t;
    }
    if (tid <= NUM_EXPERTS) offs[tid] = off_sh[tid];
}

// ---------- Kernel 2: grouped GEMM, bf16 WMMA, f32 accumulate ----------
__global__ __launch_bounds__(256)
void moe_gemm_kernel(const float* __restrict__ x, const float* __restrict__ w,
                     const int* __restrict__ perm, const int* __restrict__ offs,
                     float* __restrict__ out) {
    __shared__ unsigned short aLds[2][BM * LDSTR];
    __shared__ unsigned short bLds[2][BN * LDSTR];
    __shared__ int sTok[BM];

    // Map blockIdx.x -> (expert, m-tile) over the sorted token list (uniform).
    int rem = blockIdx.x;
    int e = -1, pStart = 0, pEnd = 0;
    #pragma unroll
    for (int i = 0; i < NUM_EXPERTS; ++i) {
        int b = offs[i], t = offs[i + 1];
        int nt = (t - b + BM - 1) / BM;
        if (e < 0) {
            if (rem < nt) { e = i; pStart = b + rem * BM; pEnd = t; }
            else rem -= nt;
        }
    }
    if (e < 0) return;  // uniform exit for unused tiles

    const int n0  = blockIdx.y * BN;
    const int tid = threadIdx.x;

    if (tid < BM) {
        int p = pStart + tid;
        sTok[tid] = (p < pEnd) ? perm[p] : -1;
    }
    __syncthreads();

    // Global-load assignment: 256 threads cover 128 rows x 8 float4 columns, 4 rows each.
    const int c  = tid & 7;    // float4 column within BK
    const int r0 = tid >> 3;   // 0..31
    const float* pa[4];
    const float* pb[4];
    #pragma unroll
    for (int s = 0; s < 4; ++s) {
        int r = r0 + 32 * s;
        int p = pStart + r;
        int tok = (p < pEnd) ? perm[p] : -1;
        pa[s] = (tok >= 0) ? (x + (size_t)tok * IN_FEAT) : (const float*)0;
        pb[s] = w + ((size_t)e * OUT_FEAT + (size_t)(n0 + r)) * IN_FEAT;
    }

    const v4f zero4 = {0.f, 0.f, 0.f, 0.f};
    v4f aR[4], bR[4];

    auto loadRegs = [&](int kt) {
        const int koff = kt * BK + c * 4;
        #pragma unroll
        for (int s = 0; s < 4; ++s) {
            aR[s] = pa[s] ? *(const v4f*)(pa[s] + koff) : zero4;
            bR[s] = *(const v4f*)(pb[s] + koff);
        }
    };
    auto storeLds = [&](int buf) {
        #pragma unroll
        for (int s = 0; s < 4; ++s) {
            const int row = r0 + 32 * s;
            v2u av; av[0] = pk2(aR[s][0], aR[s][1]); av[1] = pk2(aR[s][2], aR[s][3]);
            *(v2u*)&aLds[buf][row * LDSTR + c * 4] = av;
            v2u bv; bv[0] = pk2(bR[s][0], bR[s][1]); bv[1] = pk2(bR[s][2], bR[s][3]);
            *(v2u*)&bLds[buf][row * LDSTR + c * 4] = bv;
        }
    };

    // Wave tiling: 8 waves (2x4); each wave owns 64(M) x 32(N) of C.
    const int lane = tid & 31;
    const int wave = tid >> 5;
    const int wm   = wave >> 2;   // 0..1
    const int wn   = wave & 3;    // 0..3
    const int l    = lane & 15;
    const int half = lane >> 4;

    v8f acc[4][2] = {};

    union Frag { struct { v4u lo, hi; } u; v16bf v; };

    auto compute = [&](int buf) {
        Frag af[4], bf[2];
        // A 16x32 bf16 frag: lane<16 -> M=l, K {0..7,16..23}; lane>=16 -> K {8..15,24..31}
        #pragma unroll
        for (int i = 0; i < 4; ++i) {
            const v4u* p = (const v4u*)&aLds[buf][(wm * 64 + i * 16 + l) * LDSTR + half * 8];
            af[i].u.lo = p[0];
            af[i].u.hi = p[2];  // +32 bytes
        }
        // B 32x16 bf16 frag: lane<16 -> N=l, K {0..7,16..23}; lane>=16 -> K {8..15,24..31}
        #pragma unroll
        for (int j = 0; j < 2; ++j) {
            const v4u* q = (const v4u*)&bLds[buf][(wn * 32 + j * 16 + l) * LDSTR + half * 8];
            bf[j].u.lo = q[0];
            bf[j].u.hi = q[2];
        }
        #pragma unroll
        for (int i = 0; i < 4; ++i)
            #pragma unroll
            for (int j = 0; j < 2; ++j)
                acc[i][j] = __builtin_amdgcn_wmma_f32_16x16x32_bf16(
                    false, af[i].v, false, bf[j].v, (short)0, acc[i][j], false, false);
    };

    loadRegs(0);
    #pragma unroll 1
    for (int kt = 0; kt < KSTEPS; ++kt) {
        const int buf = kt & 1;
        storeLds(buf);
        __syncthreads();
        if (kt + 1 < KSTEPS) loadRegs(kt + 1);  // overlap next global loads with WMMA
        compute(buf);
        // single barrier per iter is safe with ping-pong buffers:
        // every thread finishes compute(kt) before the barrier in iter kt+1,
        // and storeLds(kt+2) (same buffer as compute(kt)) is after that barrier.
    }

    // Epilogue: C layout -> lane gives N=l, VGPR r gives M = half*8 + r.
    #pragma unroll
    for (int i = 0; i < 4; ++i) {
        #pragma unroll
        for (int j = 0; j < 2; ++j) {
            const int ncol = n0 + wn * 32 + j * 16 + l;
            #pragma unroll
            for (int r = 0; r < 8; ++r) {
                const int m = wm * 64 + i * 16 + half * 8 + r;
                const int tok = sTok[m];
                if (tok >= 0)
                    out[(size_t)tok * OUT_FEAT + ncol] = acc[i][j][r];
            }
        }
    }
}

extern "C" void kernel_launch(void* const* d_in, const int* in_sizes, int n_in,
                              void* d_out, int out_size, void* d_ws, size_t ws_size,
                              hipStream_t stream) {
    const float* x    = (const float*)d_in[0];
    const int*   eids = (const int*)d_in[1];
    const float* w    = (const float*)d_in[2];
    float*       out  = (float*)d_out;

    int* perm = (int*)d_ws;                 // NUM_TOKENS ints
    int* offs = perm + NUM_TOKENS;          // NUM_EXPERTS+1 ints

    moe_sort_kernel<<<1, 1024, 0, stream>>>(eids, perm, offs);

    dim3 grid(MAX_MTILES, OUT_FEAT / BN);   // 71 x 64
    moe_gemm_kernel<<<grid, 256, 0, stream>>>(x, w, perm, offs, out);
}
